// Attention_7378753814905
// MI455X (gfx1250) — matrix operand order repaired and checked by reading, not compiled, not run on hardware
//
#include <hip/hip_runtime.h>
#include <hip/hip_bf16.h>

// ---------------------------------------------------------------------------
// CDNA5 / gfx1250 wave32 WMMA implementation of NSA-style attention.
//   q,k,v = x @ w{q,k,v}^T ; RoPE(q,k) ; compressed KV = mean-pool 16;
//   joint softmax over [compressed blocks | 256-token causal window];
//   out = (P @ [cv;v]) @ wo^T
// GEMMs: bf16 WMMA with double-buffered async global->LDS staging.
// ---------------------------------------------------------------------------

typedef __attribute__((ext_vector_type(16))) __bf16 v16bf;
typedef __attribute__((ext_vector_type(8)))  float  v8f;
typedef __attribute__((ext_vector_type(4)))  int    i32x4;

union Frag16 {
    v16bf v;
    unsigned int u[8];
    uint4 q[2];
};

#define SEQ   2048
#define DMODEL 2048
#define NH    16
#define DH    128
#define NC    128      // SEQ / 16 compressed blocks
#define WIN   256
#define NEGINF (-1e30f)

#if defined(__has_builtin)
#if __has_builtin(__builtin_amdgcn_global_load_async_to_lds_b128) && \
    __has_builtin(__builtin_amdgcn_s_wait_asynccnt)
#define HAVE_ASYNC_LDS 1
#endif
#endif

__device__ __forceinline__ unsigned short f2bf(float f) {
    unsigned int u = __float_as_uint(f);
    unsigned int r = u + 0x7FFFu + ((u >> 16) & 1u);   // round-to-nearest-even
    return (unsigned short)(r >> 16);
}
__device__ __forceinline__ float bf2f(unsigned short h) {
    return __uint_as_float(((unsigned int)h) << 16);
}

// 16-bit A-matrix 16x32 layout (ISA 7.12.2): VGPR i holds K pair
__device__ __forceinline__ int akoff(int i, int half) {
    return 2 * i + ((i >= 4) ? 8 : 0) + 8 * half;
}
// 16-bit B-matrix 32x16 layout: same K pairs as A (table order)
__device__ __forceinline__ int bkoff(int i, int half) {
    return 2 * i + ((i >= 4) ? 8 : 0) + 8 * half;
}

__device__ __forceinline__ v8f wmma_bf16(const Frag16& a, const Frag16& b, v8f c) {
    return __builtin_amdgcn_wmma_f32_16x16x32_bf16(
        false, a.v, false, b.v, (short)0, c, false, false);
}

// async global -> LDS 16B copy (ASYNCcnt-tracked); sync fallback if absent
__device__ __forceinline__ void async_ld_b128(const unsigned short* g, unsigned short* l) {
#ifdef HAVE_ASYNC_LDS
    typedef __attribute__((address_space(1))) i32x4 gvec_t;
    typedef __attribute__((address_space(3))) i32x4 lvec_t;
    __builtin_amdgcn_global_load_async_to_lds_b128(
        (gvec_t*)(i32x4*)(void*)const_cast<unsigned short*>(g),
        (lvec_t*)(i32x4*)(void*)l, 0, 0);
#else
    *(uint4*)l = *(const uint4*)g;
#endif
}
#ifdef HAVE_ASYNC_LDS
#define WAIT_ASYNC(n) __builtin_amdgcn_s_wait_asynccnt(n)
#else
#define WAIT_ASYNC(n)
#endif

// ---------------------------------------------------------------------------
// elementwise: f32 -> bf16 cast
// ---------------------------------------------------------------------------
__global__ void cast_f32_bf16(const float* __restrict__ in,
                              unsigned short* __restrict__ out, int n) {
    int i = blockIdx.x * blockDim.x + threadIdx.x;
    if (i < n) out[i] = f2bf(in[i]);
}

// ---------------------------------------------------------------------------
// NT GEMM: C[M,N] = A[M,K] * B[N,K]^T, bf16 inputs, f32 output, row major.
// 256 threads = 8 waves; block tile 128x128; wave tile 32(M)x64(N).
// K-slices (128x32 A + 128x32 B, 16KB) double-buffered in LDS via
// GLOBAL_LOAD_ASYNC_TO_LDS_B128, overlapped with WMMA on the other buffer.
// ---------------------------------------------------------------------------
__global__ __launch_bounds__(256)
void gemm_nt_bf16(const unsigned short* __restrict__ A,
                  const unsigned short* __restrict__ B,
                  float* __restrict__ C, int M, int N, int K) {
    __shared__ unsigned short bufA[2][128 * 32];
    __shared__ unsigned short bufB[2][128 * 32];

    const int tid  = threadIdx.x;
    const int lane = tid & 31;
    const int wave = tid >> 5;
    const int nloc = lane & 15;
    const int half = lane >> 4;
    const int wm   = wave & 3;      // 4 M-positions * 32
    const int wn   = wave >> 2;     // 2 N-positions * 64
    const int tm0  = blockIdx.y * 128;
    const int tn0  = blockIdx.x * 128;

    // each thread stages 2 b128 per 128x32 tile (512 x 16B = 8KB)
    auto stage = [&](const unsigned short* __restrict__ G, int row0, int k0,
                     unsigned short* lbuf) {
#pragma unroll
        for (int l = 0; l < 2; ++l) {
            int lin = tid * 2 + l;          // 0..511
            int r   = lin >> 2;             // row 0..127
            int seg = lin & 3;              // 16B segment within 64B row
            async_ld_b128(G + (size_t)(row0 + r) * K + k0 + seg * 8,
                          lbuf + r * 32 + seg * 8);
        }
    };

    v8f acc[2][4] = {};
    const int nk = K / 32;

    stage(A, tm0, 0, bufA[0]);
    stage(B, tn0, 0, bufB[0]);

    for (int kt = 0; kt < nk; ++kt) {
        const int cur = kt & 1;
        if (kt + 1 < nk) {
            // previous compute's trailing barrier guarantees buffer is free
            stage(A, tm0, (kt + 1) * 32, bufA[1 - cur]);
            stage(B, tn0, (kt + 1) * 32, bufB[1 - cur]);
            WAIT_ASYNC(4);                  // 4 just-issued may remain in flight
        } else {
            WAIT_ASYNC(0);
        }
        __syncthreads();

        const unsigned short* As = bufA[cur];
        const unsigned short* Bs = bufB[cur];
        Frag16 a[2], b[4];
#pragma unroll
        for (int mt = 0; mt < 2; ++mt) {
            const unsigned short* p = As + (wm * 32 + mt * 16 + nloc) * 32 + 8 * half;
            a[mt].q[0] = *(const uint4*)(p);
            a[mt].q[1] = *(const uint4*)(p + 16);
        }
#pragma unroll
        for (int nt = 0; nt < 4; ++nt) {
            const unsigned short* p = Bs + (wn * 64 + nt * 16 + nloc) * 32 + 8 * half;
            b[nt].q[0] = *(const uint4*)(p);
            b[nt].q[1] = *(const uint4*)(p + 16);
        }
#pragma unroll
        for (int mt = 0; mt < 2; ++mt)
#pragma unroll
            for (int nt = 0; nt < 4; ++nt)
                acc[mt][nt] = wmma_bf16(a[mt], b[nt], acc[mt][nt]);
        __syncthreads();
    }

    // C/D layout: VGPR j -> row j + 8*half ; col = nloc
#pragma unroll
    for (int mt = 0; mt < 2; ++mt)
#pragma unroll
        for (int nt = 0; nt < 4; ++nt)
#pragma unroll
            for (int j = 0; j < 8; ++j)
                C[(size_t)(tm0 + wm * 32 + mt * 16 + j + 8 * half) * N +
                  (tn0 + wn * 64 + nt * 16 + nloc)] = acc[mt][nt][j];
}

// ---------------------------------------------------------------------------
// RoPE (interleaved complex pairs) + repack q,k f32 [s, h*dh] -> bf16 [h][s][dh]
// ---------------------------------------------------------------------------
__global__ void rope_pack(const float* __restrict__ qf, const float* __restrict__ kf,
                          unsigned short* __restrict__ qb, unsigned short* __restrict__ kb) {
    int t = blockIdx.x * blockDim.x + threadIdx.x;
    if (t >= SEQ * NH * (DH / 2)) return;
    int p  = t & 63;            // pair index within head dim
    int hh = (t >> 6) & (NH - 1);
    int i  = t >> 10;           // token
    float inv = __expf(-((float)p) * (9.210340371976184f / 64.0f)); // base^(-2p/dh)
    float ang = (float)i * inv;
    float sn, cs;
    __sincosf(ang, &sn, &cs);
    int src = i * DMODEL + hh * DH + 2 * p;
    int dst = (hh * SEQ + i) * DH + 2 * p;
    float qr = qf[src], qi = qf[src + 1];
    float kr = kf[src], ki = kf[src + 1];
    qb[dst]     = f2bf(qr * cs - qi * sn);
    qb[dst + 1] = f2bf(qr * sn + qi * cs);
    kb[dst]     = f2bf(kr * cs - ki * sn);
    kb[dst + 1] = f2bf(kr * sn + ki * cs);
}

// v f32 [s, h*dh] -> vt bf16 [h][dh][s]  (transposed so PV B-frags are contiguous)
__global__ void pack_vt(const float* __restrict__ vf, unsigned short* __restrict__ vt) {
    int idx = blockIdx.x * blockDim.x + threadIdx.x;
    if (idx >= NH * DH * SEQ) return;
    int j  = idx & (SEQ - 1);
    int d  = (idx >> 11) & (DH - 1);
    int hh = idx >> 18;
    vt[idx] = f2bf(vf[(size_t)j * DMODEL + hh * DH + d]);
}

// compressed K: ck[h][c][dh] = mean over 16 post-RoPE k tokens
__global__ void compress_k(const unsigned short* __restrict__ kb,
                           unsigned short* __restrict__ ck) {
    int idx = blockIdx.x * blockDim.x + threadIdx.x;
    if (idx >= NH * NC * DH) return;
    int d  = idx & (DH - 1);
    int c  = (idx >> 7) & (NC - 1);
    int hh = idx >> 14;
    float s = 0.f;
#pragma unroll
    for (int r = 0; r < 16; ++r)
        s += bf2f(kb[(size_t)(hh * SEQ + c * 16 + r) * DH + d]);
    ck[idx] = f2bf(s * (1.0f / 16.0f));
}

// compressed V transposed: cvt[h][dh][c] = mean over 16 tokens of vt
__global__ void compress_v(const unsigned short* __restrict__ vt,
                           unsigned short* __restrict__ cvt) {
    int idx = blockIdx.x * blockDim.x + threadIdx.x;
    if (idx >= NH * DH * NC) return;
    int c  = idx & (NC - 1);
    int d  = (idx >> 7) & (DH - 1);
    int hh = idx >> 14;
    float s = 0.f;
#pragma unroll
    for (int r = 0; r < 16; ++r)
        s += bf2f(vt[(size_t)(hh * DH + d) * SEQ + c * 16 + r]);
    cvt[idx] = f2bf(s * (1.0f / 16.0f));
}

// ---------------------------------------------------------------------------
// Flash attention: 1 wave per (head, 16-query block). Streams 32-key steps:
// compressed blocks first, then causal sliding window; joint online softmax.
// ---------------------------------------------------------------------------
__global__ __launch_bounds__(32)
void attn_fwd(const unsigned short* __restrict__ qb,
              const unsigned short* __restrict__ kb,
              const unsigned short* __restrict__ vt,
              const unsigned short* __restrict__ ck,
              const unsigned short* __restrict__ cvt,
              unsigned short* __restrict__ ob) {
    const int lane = threadIdx.x;
    const int nloc = lane & 15;
    const int half = lane >> 4;
    const int qblk = blockIdx.x & (SEQ / 16 - 1);
    const int h    = blockIdx.x >> 7;
    const int i0   = qblk * 16;
    const float scale = 0.08838834764831845f;   // 128^-0.5

    __shared__ float          sT[16 * 32];
    __shared__ unsigned short pT[16 * 32];
    __shared__ float          alphaS[16];
    __shared__ float          lS[16];

    // cache Q A-frags for all 4 dh-chunks of K=32
    Frag16 qf[4];
#pragma unroll
    for (int kc = 0; kc < 4; ++kc) {
        const unsigned short* Qr = qb + (size_t)(h * SEQ + i0 + nloc) * DH + kc * 32;
#pragma unroll
        for (int i = 0; i < 8; ++i)
            qf[kc].u[i] = *(const unsigned int*)(Qr + akoff(i, half));
    }

    v8f oacc[8] = {};
    float m_run = NEGINF, l_run = 0.f;

    auto step = [&](const unsigned short* Kbase, const unsigned short* Vbase,
                    int klen, int jbase, bool comp) {
        // ---- scores: 2 N-tiles x 4 dh-chunks = 8 WMMAs
        v8f sacc[2] = {};
#pragma unroll
        for (int kc = 0; kc < 4; ++kc) {
            Frag16 bf[2];
#pragma unroll
            for (int t = 0; t < 2; ++t) {
                int key = jbase + t * 16 + nloc;
                if (key < 0) key = 0;
                if (key > klen - 1) key = klen - 1;      // clamp addr; mask below
                const unsigned short* Kr = Kbase + (size_t)key * DH + kc * 32;
#pragma unroll
                for (int i = 0; i < 8; ++i)
                    bf[t].u[i] = *(const unsigned int*)(Kr + bkoff(i, half));
            }
            sacc[0] = wmma_bf16(qf[kc], bf[0], sacc[0]);
            sacc[1] = wmma_bf16(qf[kc], bf[1], sacc[1]);
        }
        // ---- scale + mask into LDS (C layout: row = j + 8*half, col = nloc)
#pragma unroll
        for (int t = 0; t < 2; ++t) {
            int col = jbase + t * 16 + nloc;
#pragma unroll
            for (int j = 0; j < 8; ++j) {
                int row = j + 8 * half;
                int qi  = i0 + row;
                float v = sacc[t][j] * scale;
                bool vis;
                if (comp) vis = (col >= 0) && (col < klen) && (col < ((qi + 1) >> 4));
                else      vis = (col >= 0) && (col < klen) && (col <= qi) && (col > qi - WIN);
                sT[row * 32 + t * 16 + nloc] = vis ? v : NEGINF;
            }
        }
        __syncthreads();
        // ---- online softmax, one lane per row
        if (lane < 16) {
            int r = lane;
            float mn = m_run;
            for (int c = 0; c < 32; ++c) mn = fmaxf(mn, sT[r * 32 + c]);
            float alpha = __expf(m_run - mn);
            float ls = 0.f;
            for (int c = 0; c < 32; ++c) {
                float p = __expf(sT[r * 32 + c] - mn);
                ls += p;
                pT[r * 32 + c] = f2bf(p);
            }
            m_run = mn;
            l_run = l_run * alpha + ls;
            alphaS[r] = alpha;
        }
        __syncthreads();
        // ---- rescale O accumulator
        float al[8];
#pragma unroll
        for (int j = 0; j < 8; ++j) al[j] = alphaS[j + 8 * half];
#pragma unroll
        for (int t = 0; t < 8; ++t)
#pragma unroll
            for (int j = 0; j < 8; ++j) oacc[t][j] *= al[j];
        // ---- P A-frag from LDS
        Frag16 pf;
#pragma unroll
        for (int i = 0; i < 8; ++i)
            pf.u[i] = *(const unsigned int*)&pT[nloc * 32 + akoff(i, half)];
        // ---- PV: 8 dh-tiles, V^T gives contiguous K-pairs
#pragma unroll
        for (int t = 0; t < 8; ++t) {
            int d = t * 16 + nloc;
            const unsigned short* Vr = Vbase + (size_t)d * klen;
            Frag16 bv;
#pragma unroll
            for (int i = 0; i < 8; ++i) {
                int src = jbase + bkoff(i, half);
                if (src > klen - 2) src = klen - 2;      // p==0 there anyway
                if (src < 0) src = 0;
                bv.u[i] = *(const unsigned int*)(Vr + src);
            }
            oacc[t] = wmma_bf16(pf, bv, oacc[t]);
        }
        __syncthreads();
    };

    // compressed branch: blocks j < (i+1)//16  (uniform loop bound per q-block)
    const unsigned short* ckH  = ck  + (size_t)h * NC * DH;
    const unsigned short* cvtH = cvt + (size_t)h * DH * NC;
    int climit = qblk + 1;                          // max visible block idx + 1
    for (int jb = 0; jb < climit; jb += 32) step(ckH, cvtH, NC, jb, true);

    // sliding window branch
    const unsigned short* kbH = kb + (size_t)h * SEQ * DH;
    const unsigned short* vtH = vt + (size_t)h * DH * SEQ;
    int lo = i0 - (WIN - 1);
    if (lo < 0) lo = 0;
    lo &= ~31;
    for (int jb = lo; jb < i0 + 16; jb += 32) step(kbH, vtH, SEQ, jb, false);

    // ---- finalize: O /= l, write bf16 o [s, h*dh]
    if (lane < 16) lS[lane] = l_run;
    __syncthreads();
    float linv[8];
#pragma unroll
    for (int j = 0; j < 8; ++j) linv[j] = 1.0f / lS[j + 8 * half];
#pragma unroll
    for (int t = 0; t < 8; ++t)
#pragma unroll
        for (int j = 0; j < 8; ++j) {
            int row = i0 + j + 8 * half;
            int col = h * DH + t * 16 + nloc;
            ob[(size_t)row * DMODEL + col] = f2bf(oacc[t][j] * linv[j]);
        }
}

// ---------------------------------------------------------------------------
// host-side orchestration
// ---------------------------------------------------------------------------
extern "C" void kernel_launch(void* const* d_in, const int* in_sizes, int n_in,
                              void* d_out, int out_size, void* d_ws, size_t ws_size,
                              hipStream_t stream) {
    const float* x  = (const float*)d_in[0];
    const float* wq = (const float*)d_in[1];
    const float* wk = (const float*)d_in[2];
    const float* wv = (const float*)d_in[3];
    const float* wo = (const float*)d_in[4];

    const size_t NE   = (size_t)SEQ * DMODEL;      // 4M elements
    unsigned char* p  = (unsigned char*)d_ws;
    auto take = [&](size_t bytes) { unsigned char* r = p; p += (bytes + 255) & ~(size_t)255; return r; };

    unsigned short* xb  = (unsigned short*)take(NE * 2);
    unsigned short* wqb = (unsigned short*)take(NE * 2);
    unsigned short* wkb = (unsigned short*)take(NE * 2);
    unsigned short* wvb = (unsigned short*)take(NE * 2);
    unsigned short* wob = (unsigned short*)take(NE * 2);
    float* qf = (float*)take(NE * 4);
    float* kf = (float*)take(NE * 4);
    float* vf = (float*)take(NE * 4);
    unsigned short* qbf = (unsigned short*)take(NE * 2);   // [h][s][dh]
    unsigned short* kbf = (unsigned short*)take(NE * 2);   // [h][s][dh]
    unsigned short* vtb = (unsigned short*)take(NE * 2);   // [h][dh][s]
    unsigned short* ckb = (unsigned short*)take((size_t)NH * NC * DH * 2);
    unsigned short* cvb = (unsigned short*)take((size_t)NH * DH * NC * 2);
    unsigned short* obf = (unsigned short*)take(NE * 2);   // [s, h*dh]

    const int TPB = 256;
    int nb = (int)((NE + TPB - 1) / TPB);
    cast_f32_bf16<<<nb, TPB, 0, stream>>>(x,  xb,  (int)NE);
    cast_f32_bf16<<<nb, TPB, 0, stream>>>(wq, wqb, (int)NE);
    cast_f32_bf16<<<nb, TPB, 0, stream>>>(wk, wkb, (int)NE);
    cast_f32_bf16<<<nb, TPB, 0, stream>>>(wv, wvb, (int)NE);
    cast_f32_bf16<<<nb, TPB, 0, stream>>>(wo, wob, (int)NE);

    dim3 ggrid(DMODEL / 128, SEQ / 128);
    gemm_nt_bf16<<<ggrid, TPB, 0, stream>>>(xb, wqb, qf, SEQ, DMODEL, DMODEL);
    gemm_nt_bf16<<<ggrid, TPB, 0, stream>>>(xb, wkb, kf, SEQ, DMODEL, DMODEL);
    gemm_nt_bf16<<<ggrid, TPB, 0, stream>>>(xb, wvb, vf, SEQ, DMODEL, DMODEL);

    int nrope = SEQ * NH * (DH / 2);
    rope_pack<<<(nrope + TPB - 1) / TPB, TPB, 0, stream>>>(qf, kf, qbf, kbf);
    pack_vt<<<(int)((NE + TPB - 1) / TPB), TPB, 0, stream>>>(vf, vtb);
    compress_k<<<(NH * NC * DH + TPB - 1) / TPB, TPB, 0, stream>>>(kbf, ckb);
    compress_v<<<(NH * DH * NC + TPB - 1) / TPB, TPB, 0, stream>>>(vtb, cvb);

    attn_fwd<<<(SEQ / 16) * NH, 32, 0, stream>>>(qbf, kbf, vtb, ckb, cvb, obf);

    gemm_nt_bf16<<<ggrid, TPB, 0, stream>>>(obf, wob, (float*)d_out, SEQ, DMODEL, DMODEL);
}
